// LONG_MIL_V1_20160576488021
// MI455X (gfx1250) — matrix operand order repaired and checked by reading, not compiled, not run on hardware
//
#include <hip/hip_runtime.h>
#include <stdint.h>

// ---------------- constants ----------------
#define N_TOK   4096
#define IN_DIM  1024
#define HID     512
#define HEADS   4
#define DH      128
#define TDIM    9332            // 31*300+31+1
#define QKV_LD  1536            // 3*HID
#define COORD_UNIT 896.0f       // 112*2*4

typedef __attribute__((ext_vector_type(16))) __bf16 v16bf;
typedef __attribute__((ext_vector_type(8)))  float  v8f;
typedef __attribute__((ext_vector_type(4)))  unsigned uint32x4;
typedef __attribute__((ext_vector_type(8)))  unsigned uint32x8;

union FragU {
  v16bf v;
  unsigned u[8];
  unsigned short s[16];
};
union G0U { uint32x4 v; unsigned w[4]; };
union G1U { uint32x8 v; unsigned w[8]; };

__device__ inline unsigned short f2bf(float f) {
  unsigned u = __float_as_uint(f);
  unsigned r = u + 0x7FFFu + ((u >> 16) & 1u);
  return (unsigned short)(r >> 16);
}

__device__ inline v8f wmma_bf16(v16bf a, v16bf b, v8f c) {
  return __builtin_amdgcn_wmma_f32_16x16x32_bf16(false, a, false, b, (short)0, c, false, false);
}

__device__ inline unsigned lds_addr_of(const void* p) {
  return (unsigned)(uintptr_t)p;   // shared aperture: low 32 bits = LDS offset
}

// CDNA5 async global->LDS copy (16B per lane), tracked by ASYNCcnt.
__device__ inline void async_b128(unsigned lds_off, const void* gaddr) {
  asm volatile("global_load_async_to_lds_b128 %0, %1, off"
               :: "v"(lds_off), "v"(gaddr) : "memory");
}
__device__ inline void wait_async0() {
  asm volatile("s_wait_asynccnt 0x0" ::: "memory");
}

// CDNA5 Tensor Data Mover: one instruction moves a 2D tile global->LDS.
// D# per CDNA5 ISA: group0 = {flags, lds_addr, global_addr, type=2},
// group1 = {mask/data_size/pad, tensor dims, tile dims, dim0 stride}. data_size=2B.
__device__ inline void tdm_load_2d(unsigned lds_off, const void* gaddr,
                                   unsigned tile_w, unsigned tile_h,
                                   unsigned stride_elems,
                                   unsigned pad_interval, unsigned pad_amount) {
  unsigned long ga = (unsigned long)gaddr;
  G0U g0; G1U g1;
  g0.w[0] = 1u;                                            // count=1, user mode
  g0.w[1] = lds_off;                                       // lds_addr
  g0.w[2] = (unsigned)ga;                                  // global_addr[31:0]
  g0.w[3] = (unsigned)((ga >> 32) & 0x01FFFFFFu) | (2u << 30); // addr[56:32], type=2
  unsigned td0 = stride_elems;                             // tensor_dim0 (elems)
  unsigned td1 = 1u << 20;                                 // tensor_dim1 (large)
  g1.w[0] = (1u << 16) | (1u << 20) | (pad_interval << 22) | (pad_amount << 25);
  g1.w[1] = (td0 & 0xFFFFu) << 16;                         // tensor_dim0[15:0]
  g1.w[2] = (td0 >> 16) | ((td1 & 0xFFFFu) << 16);         // dim0[31:16] | dim1[15:0]
  g1.w[3] = (td1 >> 16) | (tile_w << 16);                  // dim1[31:16] | tile_dim0
  g1.w[4] = tile_h;                                        // tile_dim1, tile_dim2=0
  g1.w[5] = stride_elems;                                  // tensor_dim0_stride[31:0]
  g1.w[6] = 0u;
  g1.w[7] = 0u;
  asm volatile("tensor_load_to_lds %0, %1" :: "s"(g0.v), "s"(g1.v) : "memory");
}

// ---------------- conversion kernels ----------------
__global__ void cvt_kernel(const float* __restrict__ src, unsigned short* __restrict__ dst, long n) {
  long i = (long)blockIdx.x * blockDim.x + threadIdx.x;
  long stride = (long)gridDim.x * blockDim.x;
  for (; i < n; i += stride) dst[i] = f2bf(src[i]);
}

__global__ void cvt_rows_kernel(const float* __restrict__ src, int ld,
                                unsigned short* __restrict__ dst, int cols, long n) {
  long i = (long)blockIdx.x * blockDim.x + threadIdx.x;
  long stride = (long)gridDim.x * blockDim.x;
  for (; i < n; i += stride) {
    long r = i / cols, c = i % cols;
    dst[i] = f2bf(src[r * ld + c]);
  }
}

// ---------------- coord min / idx / bias min ----------------
__global__ void coordmin_kernel(const float* __restrict__ x, float* __restrict__ gmin) {
  __shared__ float rx[256], ry[256];
  int tid = threadIdx.x;
  float mx = 1e30f, my = 1e30f;
  for (int n = tid; n < N_TOK; n += 256) {
    mx = fminf(mx, x[(long)n * (IN_DIM + 2) + IN_DIM]);
    my = fminf(my, x[(long)n * (IN_DIM + 2) + IN_DIM + 1]);
  }
  rx[tid] = mx; ry[tid] = my; __syncthreads();
  for (int off = 128; off >= 1; off >>= 1) {
    if (tid < off) { rx[tid] = fminf(rx[tid], rx[tid + off]); ry[tid] = fminf(ry[tid], ry[tid + off]); }
    __syncthreads();
  }
  if (tid == 0) { gmin[0] = rx[0]; gmin[1] = ry[0]; }
}

__global__ void idx_kernel(const float* __restrict__ x, const float* __restrict__ gmin,
                           int* __restrict__ idx) {
  int n = blockIdx.x * blockDim.x + threadIdx.x;
  if (n < N_TOK) {
    float xp = rintf((x[(long)n * (IN_DIM + 2) + IN_DIM]     - gmin[0]) / COORD_UNIT);
    float yp = rintf((x[(long)n * (IN_DIM + 2) + IN_DIM + 1] - gmin[1]) / COORD_UNIT);
    idx[n] = (int)(xp * 300.0f + yp);
  }
}

__global__ void biasmin_kernel(const float* __restrict__ alibi, const int* __restrict__ idx,
                               float* __restrict__ partial) {
  long total = (long)N_TOK * N_TOK;
  long i = (long)blockIdx.x * blockDim.x + threadIdx.x;
  long stride = (long)gridDim.x * blockDim.x;
  float mn = 1e30f;
  for (; i < total; i += stride) {
    int n = (int)(i >> 12), m = (int)(i & 4095);
    mn = fminf(mn, alibi[(long)idx[n] * TDIM + idx[m]]);
  }
  __shared__ float red[256];
  red[threadIdx.x] = mn; __syncthreads();
  for (int off = 128; off >= 1; off >>= 1) {
    if (threadIdx.x < off) red[threadIdx.x] = fminf(red[threadIdx.x], red[threadIdx.x + off]);
    __syncthreads();
  }
  if (threadIdx.x == 0) partial[blockIdx.x] = red[0];
}

__global__ void minfinal_kernel(const float* __restrict__ partial, int n, float* __restrict__ gminb) {
  __shared__ float red[256];
  float mn = 1e30f;
  for (int i = threadIdx.x; i < n; i += 256) mn = fminf(mn, partial[i]);
  red[threadIdx.x] = mn; __syncthreads();
  for (int off = 128; off >= 1; off >>= 1) {
    if (threadIdx.x < off) red[threadIdx.x] = fminf(red[threadIdx.x], red[threadIdx.x + off]);
    __syncthreads();
  }
  if (threadIdx.x == 0) gminb[0] = red[0];
}

// ---------------- layernorm ----------------
__global__ __launch_bounds__(128) void ln_kernel(const float* __restrict__ x,
                                                 const float* __restrict__ s,
                                                 const float* __restrict__ b,
                                                 float* __restrict__ o) {
  int row = blockIdx.x, tid = threadIdx.x;
  const float* xr = x + (long)row * HID;
  float v[4]; float sum = 0.f, sq = 0.f;
  for (int i = 0; i < 4; i++) { v[i] = xr[tid + i * 128]; sum += v[i]; sq += v[i] * v[i]; }
  __shared__ float r0[128], r1[128];
  r0[tid] = sum; r1[tid] = sq; __syncthreads();
  for (int off = 64; off >= 1; off >>= 1) {
    if (tid < off) { r0[tid] += r0[tid + off]; r1[tid] += r1[tid + off]; }
    __syncthreads();
  }
  float mean = r0[0] / (float)HID;
  float var  = r1[0] / (float)HID - mean * mean;
  float rstd = rsqrtf(var + 1e-5f);
  for (int i = 0; i < 4; i++) {
    int c = tid + i * 128;
    o[(long)row * HID + c] = (v[i] - mean) * rstd * s[c] + b[c];
  }
}

// ---------------- bf16 WMMA GEMM (TDM-fed) ----------------
// C[M,N] = act(A_bf16[M,K] @ B_bf16[K,N] + bias + resid)
// block: 128 threads (4 waves), tile 64x64, each wave 32x32 (2x2 WMMA), k-step 32.
// Both LDS tiles are filled by the Tensor Data Mover (one descriptor each per k-step):
//   A: 64 rows x 64B (+16B pad) -> row stride 80B (A_LDU=20 uints), pi=3, pa=3
//   B: 32 rows x 128B (+16B pad) -> row stride 144B (B_LDS=72 ushorts), pi=4, pa=3
#define A_LDU 20
#define B_LDS 72
__global__ __launch_bounds__(128) void gemm_kernel(
    const unsigned short* __restrict__ A, const unsigned short* __restrict__ B,
    const float* __restrict__ bias, const float* __restrict__ resid,
    float* __restrict__ C, int M, int N, int K, int act) {
  __shared__ __attribute__((aligned(16))) unsigned AsU[64 * A_LDU];     // 5120 B
  __shared__ __attribute__((aligned(16))) unsigned short Bs[32 * B_LDS]; // 4608 B [k][n]
  int tid = threadIdx.x;
  int lane = tid & 31, wave = tid >> 5;
  int lh = lane >> 4, ll = lane & 15;
  int wm = wave >> 1, wn = wave & 1;
  int m0 = blockIdx.y * 64, n0 = blockIdx.x * 64;
  v8f acc[2][2] = {};
  int ktiles = K >> 5;
  for (int kt = 0; kt < ktiles; kt++) {
    int k0 = kt << 5;
    if (wave == 0) {
      tdm_load_2d(lds_addr_of(AsU), &A[(long)m0 * K + k0], 32, 64, (unsigned)K, 3, 3);
      tdm_load_2d(lds_addr_of(Bs),  &B[(long)k0 * N + n0], 64, 32, (unsigned)N, 4, 3);
      __builtin_amdgcn_s_wait_tensorcnt((short)0);
    }
    if (kt + 1 < ktiles)
      __builtin_prefetch(&B[(long)(k0 + 32 + (tid & 31)) * N + n0], 0, 1);
    __syncthreads();
    v16bf af[2], bfv[2];
    for (int ai = 0; ai < 2; ai++) {
      FragU fu;
      int row = wm * 32 + ai * 16 + ll;
      for (int i = 0; i < 8; i++) {
        int u = i + lh * 8;                     // k=e+lh*16, pair->uint
        fu.u[i] = AsU[row * A_LDU + u];
      }
      af[ai] = fu.v;
    }
    for (int bi = 0; bi < 2; bi++) {
      FragU fu;
      int col = wn * 32 + bi * 16 + ll;
      for (int e = 0; e < 16; e++)              // k = e + lh*16 (row of Bs)
        fu.s[e] = Bs[(e + lh * 16) * B_LDS + col];
      bfv[bi] = fu.v;
    }
    for (int ai = 0; ai < 2; ai++)
      for (int bi = 0; bi < 2; bi++)
        acc[ai][bi] = wmma_bf16(af[ai], bfv[bi], acc[ai][bi]);
    __syncthreads();
  }
  for (int ai = 0; ai < 2; ai++)
    for (int bi = 0; bi < 2; bi++)
      for (int r = 0; r < 8; r++) {
        int grow = m0 + wm * 32 + ai * 16 + r + lh * 8;
        int gcol = n0 + wn * 32 + bi * 16 + ll;
        float v = acc[ai][bi][r];
        if (bias)  v += bias[gcol];
        if (resid) v += resid[(long)grow * N + gcol];
        if (act == 1)      v = fmaxf(v, 0.f);
        else if (act == 2) v = 0.5f * v * (1.f + erff(v * 0.70710678118654752f));
        C[(long)grow * N + gcol] = v;
      }
}

// ---------------- flash attention (one wave per head x 16-query block) ----------------
#define KT_LD 136   // ushorts per row (128 data + 8 pad) -> 16B aligned chunks
__global__ __launch_bounds__(32) void attn_kernel(
    const unsigned short* __restrict__ qkv,   // [N][1536] bf16: q|k|v, each [H][DH]
    const float* __restrict__ alibi,          // [T][T]
    const int* __restrict__ gidx,             // [N]
    const float* __restrict__ gminb,          // [1]
    float* __restrict__ out,                  // [N][HID]
    int masked) {
  int head = blockIdx.x & 3;
  int q0   = (blockIdx.x >> 2) * 16;
  int lane = threadIdx.x;
  int lh = lane >> 4, ll = lane & 15;
  float slope = 0.25f / (float)(1 << (2 * head));
  const float scale = 0.08838834764831845f;   // 1/sqrt(128)
  float minb = gminb[0];

  __shared__ __attribute__((aligned(16))) unsigned short Kt[32 * KT_LD]; // [key][d]
  __shared__ __attribute__((aligned(16))) unsigned short Vt[32 * KT_LD]; // [key][d]
  __shared__ __attribute__((aligned(16))) unsigned short Pst[16 * 36];   // [row][k]
  __shared__ int idq[16];
  __shared__ int idk[32];

  if (lane < 16) idq[lane] = gidx[q0 + lane];
  __syncthreads();

  const unsigned* q32 = (const unsigned*)qkv;
  // Q fragments: 4 k-steps of 32 over DH=128
  v16bf qf[4];
  {
    long qb = ((long)(q0 + ll) * QKV_LD + head * DH) >> 1;  // uint offset
    for (int jk = 0; jk < 4; jk++) {
      FragU fu;
      for (int i = 0; i < 8; i++) {
        int u = i + lh * 8;
        fu.u[i] = q32[qb + 16 * jk + u];
      }
      qf[jk] = fu.v;
    }
  }

  v8f oacc[8] = {};
  float mrow[8], lrow[8];
  for (int r = 0; r < 8; r++) { mrow[r] = -1e30f; lrow[r] = 0.f; }

  for (int kb = 0; kb < N_TOK / 32; kb++) {
    int key0 = kb * 32;
    idk[lane] = gidx[key0 + lane];
    // K/V tiles: 32 rows x 128 bf16 each -> 512 chunks of 16B per tile, async to LDS
    for (int i = 0; i < 16; i++) {
      int e = i * 32 + lane;
      int r = e >> 4, c = e & 15;
      const unsigned short* gk = qkv + (size_t)(key0 + r) * QKV_LD + HID + head * DH + c * 8;
      async_b128(lds_addr_of(&Kt[r * KT_LD + c * 8]), gk);
      async_b128(lds_addr_of(&Vt[r * KT_LD + c * 8]), gk + HID);
    }
    wait_async0();
    __syncthreads();

    // scores: S[16q x 32k] as two 16x16 frags (B = K^T, k-dim contiguous in Kt rows)
    v8f s0 = {}, s1 = {};
    for (int jk = 0; jk < 4; jk++) {
      FragU b0, b1;
      for (int i = 0; i < 8; i++) {
        int du = 16 * jk + i + lh * 8;         // uint offset in d
        b0.u[i] = *(const unsigned*)&Kt[ll * KT_LD + 2 * du];
        b1.u[i] = *(const unsigned*)&Kt[(16 + ll) * KT_LD + 2 * du];
      }
      s0 = wmma_bf16(qf[jk], b0.v, s0);
      s1 = wmma_bf16(qf[jk], b1.v, s1);
    }

    // bias + mask + online softmax
    float sv0[8], sv1[8];
    for (int r = 0; r < 8; r++) {
      int row = r + lh * 8;
      long arow = (long)idq[row] * TDIM;
      float b0 = alibi[arow + idk[ll]];
      float b1 = alibi[arow + idk[16 + ll]];
      float x0 = s0[r] * scale + slope * b0;
      float x1 = s1[r] * scale + slope * b1;
      if (masked && head == 0) {               // bias_h==global min only at head0, b==minb
        if (b0 == minb) x0 = -1e30f;
        if (b1 == minb) x1 = -1e30f;
      }
      sv0[r] = x0; sv1[r] = x1;
    }
    for (int r = 0; r < 8; r++) {
      float v = fmaxf(sv0[r], sv1[r]);
      for (int off = 1; off < 16; off <<= 1) v = fmaxf(v, __shfl_xor(v, off, 32));
      float mnew = fmaxf(mrow[r], v);
      float corr = __expf(mrow[r] - mnew);
      float p0 = __expf(sv0[r] - mnew);
      float p1 = __expf(sv1[r] - mnew);
      float ps = p0 + p1;
      for (int off = 1; off < 16; off <<= 1) ps += __shfl_xor(ps, off, 32);
      lrow[r] = lrow[r] * corr + ps;
      mrow[r] = mnew;
      for (int j = 0; j < 8; j++) oacc[j][r] *= corr;
      sv0[r] = p0; sv1[r] = p1;
    }

    // re-layout P through LDS into an A fragment (16x32, K=keys)
    for (int r = 0; r < 8; r++) {
      int row = r + lh * 8;
      Pst[row * 36 + ll]      = f2bf(sv0[r]);
      Pst[row * 36 + 16 + ll] = f2bf(sv1[r]);
    }
    __syncthreads();
    FragU pu;
    for (int i = 0; i < 8; i++) {
      int kps = 2 * i + lh * 16;
      pu.u[i] = *(const unsigned*)&Pst[ll * 36 + kps];
    }

    // PV: 8 output col-tiles of 16 over DH; B frag element e: key=e+lh*16, col=16j+ll
    for (int j = 0; j < 8; j++) {
      FragU bu;
      int dh = 16 * j + ll;
      for (int e = 0; e < 16; e++)
        bu.s[e] = Vt[(e + lh * 16) * KT_LD + dh];
      oacc[j] = wmma_bf16(pu.v, bu.v, oacc[j]);
    }
    __syncthreads();
  }

  for (int j = 0; j < 8; j++)
    for (int r = 0; r < 8; r++) {
      int row = q0 + r + lh * 8;
      out[(long)row * HID + head * DH + 16 * j + ll] = oacc[j][r] / lrow[r];
    }
}

// ---------------- epilogue ----------------
__global__ void colmean_kernel(const float* __restrict__ h, float* __restrict__ hm) {
  int c = threadIdx.x;                         // 512 threads
  float s = 0.f;
  for (int n = 0; n < N_TOK; n++) s += h[(long)n * HID + c];
  hm[c] = s / (float)N_TOK;
}

__global__ void head_kernel(const float* __restrict__ hm, const float* __restrict__ ns,
                            const float* __restrict__ nb, const float* __restrict__ hw,
                            const float* __restrict__ hb, float* __restrict__ outp) {
  __shared__ float red[512];
  int tid = threadIdx.x;                       // 512 threads
  float v = hm[tid];
  red[tid] = v; __syncthreads();
  for (int off = 256; off >= 1; off >>= 1) { if (tid < off) red[tid] += red[tid + off]; __syncthreads(); }
  float mean = red[0] / (float)HID; __syncthreads();
  float d = v - mean;
  red[tid] = d * d; __syncthreads();
  for (int off = 256; off >= 1; off >>= 1) { if (tid < off) red[tid] += red[tid + off]; __syncthreads(); }
  float rstd = rsqrtf(red[0] / (float)HID + 1e-5f); __syncthreads();
  float hn = d * rstd * ns[tid] + nb[tid];
  for (int j = 0; j < 2; j++) {
    red[tid] = hn * hw[tid * 2 + j]; __syncthreads();
    for (int off = 256; off >= 1; off >>= 1) { if (tid < off) red[tid] += red[tid + off]; __syncthreads(); }
    if (tid == 0) outp[j] = red[0] + hb[j];
    __syncthreads();
  }
}

// ---------------- host launcher ----------------
extern "C" void kernel_launch(void* const* d_in, const int* in_sizes, int n_in,
                              void* d_out, int out_size, void* d_ws, size_t ws_size,
                              hipStream_t stream) {
  (void)in_sizes; (void)n_in; (void)out_size; (void)ws_size;
  const float* x      = (const float*)d_in[0];
  const float* alibi  = (const float*)d_in[1];
  const float* fc1_w  = (const float*)d_in[2];
  const float* fc1_b  = (const float*)d_in[3];
  const float* ln1_s  = (const float*)d_in[4];
  const float* ln1_b  = (const float*)d_in[5];
  const float* qkv_w  = (const float*)d_in[6];
  const float* proj_w = (const float*)d_in[7];
  const float* proj_b = (const float*)d_in[8];
  const float* ln2_s  = (const float*)d_in[9];
  const float* ln2_b  = (const float*)d_in[10];
  const float* mlp1_w = (const float*)d_in[11];
  const float* mlp1_b = (const float*)d_in[12];
  const float* mlp2_w = (const float*)d_in[13];
  const float* mlp2_b = (const float*)d_in[14];
  const float* norm_s = (const float*)d_in[15];
  const float* norm_b = (const float*)d_in[16];
  const float* head_w = (const float*)d_in[17];
  const float* head_b = (const float*)d_in[18];

  char* ws = (char*)d_ws;
  size_t off = 0;
  auto alloc = [&](size_t bytes) -> char* {
    char* p = ws + off;
    off = (off + bytes + 255) & ~(size_t)255;
    return p;
  };
  float* h              = (float*)alloc((size_t)N_TOK * HID * 4);        // 8 MB
  float* bufB           = (float*)alloc((size_t)N_TOK * 2048 * 4);       // 32 MB
  float* bufC           = (float*)alloc((size_t)N_TOK * HID * 4);        // 8 MB
  unsigned short* bufD  = (unsigned short*)alloc((size_t)N_TOK * 2048 * 2); // 16 MB
  unsigned short* w_fc1 = (unsigned short*)alloc((size_t)IN_DIM * HID * 2);
  unsigned short* w_qkv = (unsigned short*)alloc((size_t)3 * HID * QKV_LD * 2);
  unsigned short* w_prj = (unsigned short*)alloc((size_t)3 * HID * HID * 2);
  unsigned short* w_m1  = (unsigned short*)alloc((size_t)3 * HID * 2048 * 2);
  unsigned short* w_m2  = (unsigned short*)alloc((size_t)3 * 2048 * HID * 2);
  int*   idx     = (int*)alloc(N_TOK * 4);
  float* gmin    = (float*)alloc(256);
  float* gminb   = (float*)alloc(256);
  float* partial = (float*)alloc(512 * 4);
  float* hm      = (float*)alloc(512 * 4);

  // bf16 conversions: x (strip coord cols) + all weights
  cvt_rows_kernel<<<2048, 256, 0, stream>>>(x, IN_DIM + 2, bufD, IN_DIM, (long)N_TOK * IN_DIM);
  cvt_kernel<<<256,  256, 0, stream>>>(fc1_w,  w_fc1, (long)IN_DIM * HID);
  cvt_kernel<<<1024, 256, 0, stream>>>(qkv_w,  w_qkv, (long)3 * HID * QKV_LD);
  cvt_kernel<<<512,  256, 0, stream>>>(proj_w, w_prj, (long)3 * HID * HID);
  cvt_kernel<<<1024, 256, 0, stream>>>(mlp1_w, w_m1,  (long)3 * HID * 2048);
  cvt_kernel<<<1024, 256, 0, stream>>>(mlp2_w, w_m2,  (long)3 * 2048 * HID);

  // grid index + global bias min
  coordmin_kernel<<<1, 256, 0, stream>>>(x, gmin);
  idx_kernel<<<16, 256, 0, stream>>>(x, gmin, idx);
  biasmin_kernel<<<512, 256, 0, stream>>>(alibi, idx, partial);
  minfinal_kernel<<<1, 256, 0, stream>>>(partial, 512, gminb);

  // h = relu(x @ fc1_w + fc1_b)
  gemm_kernel<<<dim3(HID / 64, N_TOK / 64), 128, 0, stream>>>(
      bufD, w_fc1, fc1_b, nullptr, h, N_TOK, HID, IN_DIM, 1);

  for (int l = 0; l < 3; l++) {
    // xn = LN1(h)
    ln_kernel<<<N_TOK, 128, 0, stream>>>(h, ln1_s + l * HID, ln1_b + l * HID, bufC);
    cvt_kernel<<<2048, 256, 0, stream>>>(bufC, bufD, (long)N_TOK * HID);
    // qkv = xn @ qkv_w[l]
    gemm_kernel<<<dim3(QKV_LD / 64, N_TOK / 64), 128, 0, stream>>>(
        bufD, w_qkv + (size_t)l * HID * QKV_LD, nullptr, nullptr, bufB, N_TOK, QKV_LD, HID, 0);
    cvt_kernel<<<2048, 256, 0, stream>>>(bufB, bufD, (long)N_TOK * QKV_LD);
    // attention -> bufC
    attn_kernel<<<HEADS * (N_TOK / 16), 32, 0, stream>>>(
        bufD, alibi, idx, gminb, bufC, (l < 2) ? 1 : 0);
    cvt_kernel<<<2048, 256, 0, stream>>>(bufC, bufD, (long)N_TOK * HID);
    // h = h + attn @ proj_w[l] + proj_b[l]
    gemm_kernel<<<dim3(HID / 64, N_TOK / 64), 128, 0, stream>>>(
        bufD, w_prj + (size_t)l * HID * HID, proj_b + l * HID, h, h, N_TOK, HID, HID, 0);
    // xn2 = LN2(h)
    ln_kernel<<<N_TOK, 128, 0, stream>>>(h, ln2_s + l * HID, ln2_b + l * HID, bufC);
    cvt_kernel<<<2048, 256, 0, stream>>>(bufC, bufD, (long)N_TOK * HID);
    // mid = gelu(xn2 @ mlp1_w[l] + mlp1_b[l])
    gemm_kernel<<<dim3(2048 / 64, N_TOK / 64), 128, 0, stream>>>(
        bufD, w_m1 + (size_t)l * HID * 2048, mlp1_b + l * 2048, nullptr, bufB, N_TOK, 2048, HID, 2);
    cvt_kernel<<<2048, 256, 0, stream>>>(bufB, bufD, (long)N_TOK * 2048);
    // h = h + mid @ mlp2_w[l] + mlp2_b[l]
    gemm_kernel<<<dim3(HID / 64, N_TOK / 64), 128, 0, stream>>>(
        bufD, w_m2 + (size_t)l * 2048 * HID, mlp2_b + l * HID, h, h, N_TOK, HID, 2048, 0);
  }

  colmean_kernel<<<1, 512, 0, stream>>>(h, hm);
  head_kernel<<<1, 512, 0, stream>>>(hm, norm_s, norm_b, head_w, head_b, (float*)d_out);
}
